// Attention3D_90331752169560
// MI455X (gfx1250) — matrix operand order repaired and checked by reading, not compiled, not run on hardware
//
#include <hip/hip_runtime.h>
#include <hip/hip_bf16.h>

typedef __bf16 v16bf __attribute__((ext_vector_type(16)));
typedef __bf16 v8bf  __attribute__((ext_vector_type(8)));
typedef float  v8f   __attribute__((ext_vector_type(8)));

namespace {
constexpr int kB    = 32;
constexpr int kN    = 513;
constexpr int kDim  = 768;
constexpr int kH    = 12;
constexpr int kHD   = 64;
constexpr int kNPad = 544;              // 17 * 32
constexpr int kRows = kB * kN;          // 16416 = 513 * 32
constexpr float kScale = 0.125f;        // 64^-0.5
}

__device__ __forceinline__ v16bf bcat(v8bf lo, v8bf hi) {
  v16bf r;
#pragma unroll
  for (int i = 0; i < 8; ++i) { r[i] = lo[i]; r[i + 8] = hi[i]; }
  return r;
}

__device__ __forceinline__ v8f wmma_bf16(v16bf a, v16bf b, v8f c) {
  return __builtin_amdgcn_wmma_f32_16x16x32_bf16(false, a, false, b, (short)0, c,
                                                 false, false);
}

__global__ void cvt_bf16(const float* __restrict__ in, __bf16* __restrict__ out, int n) {
  int i = blockIdx.x * blockDim.x + threadIdx.x;
  if (i < n) out[i] = (__bf16)in[i];
}

// Precompute bias_full[h][row(544)][col(544)]: gathered rel-pos bias with the
// key-padding mask (-1e30) baked into cols >= 513. Batch-independent: done once.
__global__ void build_bias(const float* __restrict__ bias_table,
                           const int* __restrict__ rel_index,
                           float* __restrict__ bias_full) {
  int i = blockIdx.x * blockDim.x + threadIdx.x;
  int col = i % kNPad;
  int t = i / kNPad;
  int row = t % kNPad;
  int h = t / kNPad;
  float v = 0.f;
  if (col >= kN) v = -1e30f;
  else if (row < kN) v = bias_table[rel_index[row * kN + col] * kH + h];
  bias_full[i] = v;
}

// Zero the padded rows (513..543) of q/k/v so first-run garbage can never
// poison the WMMA (0 x Inf -> NaN per ISA).
__global__ void zero_pad(__bf16* __restrict__ qws, __bf16* __restrict__ kws,
                         __bf16* __restrict__ vws) {
  const int padN = kNPad - kN;  // 31
  int i = blockIdx.x * blockDim.x + threadIdx.x;
  int total = kB * kH * padN * kHD;
  if (i >= total) return;
  int d = i % kHD;
  int t = i / kHD;
  int n = kN + (t % padN);
  int bh = t / padN;
  qws[((size_t)bh * kNPad + n) * kHD + d] = (__bf16)0.f;
  kws[((size_t)bh * kNPad + n) * kHD + d] = (__bf16)0.f;
  vws[((size_t)bh * kHD + d) * kNPad + n] = (__bf16)0.f;
}

// ---------------- QKV GEMM: 32x64 tile per wave, scatter into q/k/v ----------------
__global__ __launch_bounds__(256) void qkv_gemm(
    const __bf16* __restrict__ xb, const __bf16* __restrict__ wb,
    __bf16* __restrict__ qws, __bf16* __restrict__ kws, __bf16* __restrict__ vws) {
  const int colg = (3 * kDim) / 64;  // 36
  int wave = (int)((blockIdx.x * 256 + threadIdx.x) >> 5);
  if (wave >= (kRows / 32) * colg) return;
  int lane = threadIdx.x & 31, lg = lane >> 4, ln = lane & 15;
  int rt = wave / colg, cg = wave % colg;
  int rowbase = rt * 32, colbase = cg * 64;

  v8f acc[2][4] = {{v8f{}, v8f{}, v8f{}, v8f{}}, {v8f{}, v8f{}, v8f{}, v8f{}}};
  const __bf16* aptr0 = xb + (size_t)(rowbase + ln) * kDim + lg * 8;
  const __bf16* aptr1 = aptr0 + (size_t)16 * kDim;
#pragma unroll 4
  for (int kk = 0; kk < kDim; kk += 32) {
    v16bf a0 = bcat(*(const v8bf*)(aptr0 + kk), *(const v8bf*)(aptr0 + kk + 16));
    v16bf a1 = bcat(*(const v8bf*)(aptr1 + kk), *(const v8bf*)(aptr1 + kk + 16));
#pragma unroll
    for (int t = 0; t < 4; ++t) {
      const __bf16* bp = wb + (size_t)(colbase + t * 16 + ln) * kDim + kk + lg * 8;
      v16bf bfrag = bcat(*(const v8bf*)bp, *(const v8bf*)(bp + 16));
      acc[0][t] = wmma_bf16(a0, bfrag, acc[0][t]);
      acc[1][t] = wmma_bf16(a1, bfrag, acc[1][t]);
    }
  }
#pragma unroll
  for (int u = 0; u < 2; ++u)
#pragma unroll
    for (int t = 0; t < 4; ++t)
#pragma unroll
      for (int r = 0; r < 8; ++r) {
        int gm = rowbase + u * 16 + r + 8 * lg;
        int bb = gm / kN, nn = gm - bb * kN;
        int o = colbase + t * 16 + ln;
        int which = o / kDim, rem = o - which * kDim;
        int h = rem >> 6, d = rem & 63;
        float val = acc[u][t][r];
        size_t bh = (size_t)(bb * kH + h);
        if (which == 0)      qws[(bh * kNPad + nn) * kHD + d] = (__bf16)(val * kScale);
        else if (which == 1) kws[(bh * kNPad + nn) * kHD + d] = (__bf16)val;
        else                 vws[(bh * kHD + d) * kNPad + nn] = (__bf16)val;
      }
}

// ---------------- Fused flash attention, 32 query rows per wave --------------------
__global__ __launch_bounds__(256) void attn_fused(
    const __bf16* __restrict__ qg, const __bf16* __restrict__ kg,
    const __bf16* __restrict__ vg, const float* __restrict__ bias_full,
    __bf16* __restrict__ attout) {
  __shared__ alignas(16) __bf16 lds[8][32 * 32];  // per-wave P tile (32x32 bf16)
  const int QT = 17;                              // 32-row query tiles
  int wv = threadIdx.x >> 5;
  int gw = (int)(blockIdx.x * 8 + wv);
  int lane = threadIdx.x & 31, lg = lane >> 4, ln = lane & 15;
  int qt = gw % QT, bh = gw / QT;
  int h = bh % kH, b = bh / kH;
  int qbase = qt * 32;

  const __bf16* qb = qg + (size_t)bh * kNPad * kHD;
  const __bf16* kb = kg + (size_t)bh * kNPad * kHD;
  const __bf16* vb = vg + (size_t)bh * kHD * kNPad;
  const float* bias = bias_full + (size_t)h * kNPad * kNPad;

  v16bf aq[2][2];
#pragma unroll
  for (int u = 0; u < 2; ++u) {
    const __bf16* p = qb + (size_t)(qbase + u * 16 + ln) * kHD + lg * 8;
    aq[u][0] = bcat(*(const v8bf*)p,        *(const v8bf*)(p + 16));
    aq[u][1] = bcat(*(const v8bf*)(p + 32), *(const v8bf*)(p + 48));
  }

  float mstate[2][8], lstate[2][8];
#pragma unroll
  for (int u = 0; u < 2; ++u)
#pragma unroll
    for (int r = 0; r < 8; ++r) { mstate[u][r] = -1e30f; lstate[u][r] = 0.f; }
  v8f oacc[2][4] = {{v8f{}, v8f{}, v8f{}, v8f{}}, {v8f{}, v8f{}, v8f{}, v8f{}}};
  __bf16* lp = &lds[wv][0];

  for (int kt = 0; kt < kNPad / 32; ++kt) {  // 17 key steps of 32
    int k0 = kt * 32;
    v8f s[2][2];
#pragma unroll
    for (int c = 0; c < 2; ++c) {
      const __bf16* p = kb + (size_t)(k0 + c * 16 + ln) * kHD + lg * 8;
      v16bf b0 = bcat(*(const v8bf*)p,        *(const v8bf*)(p + 16));
      v16bf b1 = bcat(*(const v8bf*)(p + 32), *(const v8bf*)(p + 48));
#pragma unroll
      for (int u = 0; u < 2; ++u) {
        v8f cf = {};
        cf = wmma_bf16(aq[u][0], b0, cf);
        cf = wmma_bf16(aq[u][1], b1, cf);
        s[u][c] = cf;
      }
    }
    // precomputed bias (mask baked into padded cols): coalesced fp32 loads
#pragma unroll
    for (int u = 0; u < 2; ++u)
#pragma unroll
      for (int c = 0; c < 2; ++c) {
        int col = k0 + c * 16 + ln;
#pragma unroll
        for (int r = 0; r < 8; ++r) {
          int row = qbase + u * 16 + r + 8 * lg;
          s[u][c][r] += bias[(size_t)row * kNPad + col];
        }
      }
    // online softmax + P relayout through LDS
#pragma unroll
    for (int u = 0; u < 2; ++u) {
      float alpha[8];
#pragma unroll
      for (int r = 0; r < 8; ++r) {
        float mt = fmaxf(s[u][0][r], s[u][1][r]);
        mt = fmaxf(mt, __shfl_xor(mt, 1));
        mt = fmaxf(mt, __shfl_xor(mt, 2));
        mt = fmaxf(mt, __shfl_xor(mt, 4));
        mt = fmaxf(mt, __shfl_xor(mt, 8));
        float mn = fmaxf(mstate[u][r], mt);
        alpha[r] = __expf(mstate[u][r] - mn);
        float p0 = __expf(s[u][0][r] - mn);
        float p1 = __expf(s[u][1][r] - mn);
        s[u][0][r] = p0; s[u][1][r] = p1;
        float rs = p0 + p1;
        rs += __shfl_xor(rs, 1);
        rs += __shfl_xor(rs, 2);
        rs += __shfl_xor(rs, 4);
        rs += __shfl_xor(rs, 8);
        lstate[u][r] = lstate[u][r] * alpha[r] + rs;
        mstate[u][r] = mn;
      }
#pragma unroll
      for (int t = 0; t < 4; ++t)
#pragma unroll
        for (int r = 0; r < 8; ++r) oacc[u][t][r] *= alpha[r];
#pragma unroll
      for (int r = 0; r < 8; ++r) {
        int row = u * 16 + r + 8 * lg;
        lp[row * 32 + ln]      = (__bf16)s[u][0][r];
        lp[row * 32 + 16 + ln] = (__bf16)s[u][1][r];
      }
    }
    v16bf ap[2];
#pragma unroll
    for (int u = 0; u < 2; ++u) {
      const __bf16* p = lp + (u * 16 + ln) * 32 + lg * 8;
      ap[u] = bcat(*(const v8bf*)p, *(const v8bf*)(p + 16));
    }
    // O += P @ V; V fragments shared by both query halves
#pragma unroll
    for (int t = 0; t < 4; ++t) {
      const __bf16* p = vb + (size_t)(t * 16 + ln) * kNPad + k0 + lg * 8;
      v16bf bv = bcat(*(const v8bf*)p, *(const v8bf*)(p + 16));
      oacc[0][t] = wmma_bf16(ap[0], bv, oacc[0][t]);
      oacc[1][t] = wmma_bf16(ap[1], bv, oacc[1][t]);
    }
  }
#pragma unroll
  for (int u = 0; u < 2; ++u)
#pragma unroll
    for (int t = 0; t < 4; ++t)
#pragma unroll
      for (int r = 0; r < 8; ++r) {
        int qn = qbase + u * 16 + r + 8 * lg;
        if (qn < kN) {
          float val = oacc[u][t][r] / lstate[u][r];
          attout[((size_t)(b * kN + qn)) * kDim + h * kHD + t * 16 + ln] = (__bf16)val;
        }
      }
}

// ---------------- Output projection: 32x64 tile per wave, fp32 out + bias ----------
__global__ __launch_bounds__(256) void proj_gemm(
    const __bf16* __restrict__ ab, const __bf16* __restrict__ wb,
    const float* __restrict__ bias, float* __restrict__ out) {
  const int colg = kDim / 64;  // 12
  int wave = (int)((blockIdx.x * 256 + threadIdx.x) >> 5);
  if (wave >= (kRows / 32) * colg) return;
  int lane = threadIdx.x & 31, lg = lane >> 4, ln = lane & 15;
  int rt = wave / colg, cg = wave % colg;
  int rowbase = rt * 32, colbase = cg * 64;

  v8f acc[2][4] = {{v8f{}, v8f{}, v8f{}, v8f{}}, {v8f{}, v8f{}, v8f{}, v8f{}}};
  const __bf16* aptr0 = ab + (size_t)(rowbase + ln) * kDim + lg * 8;
  const __bf16* aptr1 = aptr0 + (size_t)16 * kDim;
#pragma unroll 4
  for (int kk = 0; kk < kDim; kk += 32) {
    v16bf a0 = bcat(*(const v8bf*)(aptr0 + kk), *(const v8bf*)(aptr0 + kk + 16));
    v16bf a1 = bcat(*(const v8bf*)(aptr1 + kk), *(const v8bf*)(aptr1 + kk + 16));
#pragma unroll
    for (int t = 0; t < 4; ++t) {
      const __bf16* bp = wb + (size_t)(colbase + t * 16 + ln) * kDim + kk + lg * 8;
      v16bf bfrag = bcat(*(const v8bf*)bp, *(const v8bf*)(bp + 16));
      acc[0][t] = wmma_bf16(a0, bfrag, acc[0][t]);
      acc[1][t] = wmma_bf16(a1, bfrag, acc[1][t]);
    }
  }
#pragma unroll
  for (int u = 0; u < 2; ++u)
#pragma unroll
    for (int t = 0; t < 4; ++t)
#pragma unroll
      for (int r = 0; r < 8; ++r) {
        int gm = rowbase + u * 16 + r + 8 * lg;
        int o = colbase + t * 16 + ln;
        out[(size_t)gm * kDim + o] = acc[u][t][r] + bias[o];
      }
}

extern "C" void kernel_launch(void* const* d_in, const int* in_sizes, int n_in,
                              void* d_out, int out_size, void* d_ws, size_t ws_size,
                              hipStream_t stream) {
  (void)in_sizes; (void)n_in; (void)out_size; (void)ws_size;
  const float* x          = (const float*)d_in[0];
  const float* w_qkv      = (const float*)d_in[1];
  const float* bias_table = (const float*)d_in[2];
  const float* w_proj     = (const float*)d_in[3];
  const float* b_proj     = (const float*)d_in[4];
  const int*   rel_index  = (const int*)d_in[5];
  float* out = (float*)d_out;

  char* ws = (char*)d_ws;
  size_t off = 0;
  auto take = [&](size_t bytes) -> void* {
    void* p = ws + off;
    off = (off + bytes + 255) & ~(size_t)255;
    return p;
  };
  __bf16* xb       = (__bf16*)take((size_t)kRows * kDim * 2);
  __bf16* wqkvb    = (__bf16*)take((size_t)3 * kDim * kDim * 2);
  __bf16* wprojb   = (__bf16*)take((size_t)kDim * kDim * 2);
  __bf16* qws      = (__bf16*)take((size_t)kB * kH * kNPad * kHD * 2);
  __bf16* kws      = (__bf16*)take((size_t)kB * kH * kNPad * kHD * 2);
  __bf16* vws      = (__bf16*)take((size_t)kB * kH * kHD * kNPad * 2);
  float*  biasfull = (float*)take((size_t)kH * kNPad * kNPad * 4);
  __bf16* attout   = xb;  // xb dead after qkv_gemm; reuse

  int nx = kRows * kDim;
  cvt_bf16<<<(nx + 255) / 256, 256, 0, stream>>>(x, xb, nx);
  int nw1 = 3 * kDim * kDim;
  cvt_bf16<<<(nw1 + 255) / 256, 256, 0, stream>>>(w_qkv, wqkvb, nw1);
  int nw2 = kDim * kDim;
  cvt_bf16<<<(nw2 + 255) / 256, 256, 0, stream>>>(w_proj, wprojb, nw2);

  // 12 * 544 * 544 = 3,551,232 elements -> 13872 blocks (exact)
  build_bias<<<(kH * kNPad * kNPad) / 256, 256, 0, stream>>>(bias_table, rel_index,
                                                             biasfull);
  int npad = kB * kH * (kNPad - kN) * kHD;  // 761856
  zero_pad<<<(npad + 255) / 256, 256, 0, stream>>>(qws, kws, vws);

  // 513 row tiles * 36 col groups = 18468 waves -> 2309 blocks (guarded)
  qkv_gemm<<<2309, 256, 0, stream>>>(xb, wqkvb, qws, kws, vws);
  // 32*12*17 = 6528 waves -> 816 blocks (exact)
  attn_fused<<<816, 256, 0, stream>>>(qws, kws, vws, biasfull, attout);
  // 513 * 12 = 6156 waves -> 770 blocks (guarded)
  proj_gemm<<<770, 256, 0, stream>>>(attout, wprojb, b_proj, out);
}
